// MutiHeadSelfAttention_81595788689494
// MI455X (gfx1250) — compile-verified, matrix operand order checked by reading, not run on hardware
//
#include <hip/hip_runtime.h>
#include <hip/hip_bf16.h>

// MI455X / gfx1250, wave32. bf16 WMMA pipeline for fused MHA forward.
// Projection GEMMs: 128x128 block tile, 8 waves (2/SIMD32), TDM-staged
// double-buffered LDS (s_wait_tensorcnt + barrier handshake), 2x4 WMMA per wave.
// Workspace (bf16 elems): x(4M) | Wq,Wk,Wv,Wo(1M each) | Q,K,V(4M each) | merged(4M) = 48 MB.

#define D_MODEL   1024
#define NUM_HEADS 16
#define DK        64
#define BB        2
#define TT        2048
#define MROWS     (BB * TT) /* 4096 */

typedef __attribute__((ext_vector_type(16))) __bf16 v16bf;
typedef __attribute__((ext_vector_type(8)))  __bf16 bf16x8;
typedef __attribute__((ext_vector_type(4)))  __bf16 bf16x4;
typedef __attribute__((ext_vector_type(8)))  float  v8f;
typedef __attribute__((ext_vector_type(4)))  unsigned int v4u;
typedef __attribute__((ext_vector_type(8)))  int v8i;
typedef __attribute__((ext_vector_type(4)))  int v4i;

#if defined(__has_builtin)
#if __has_builtin(__builtin_amdgcn_tensor_load_to_lds) && __has_builtin(__builtin_amdgcn_s_wait_tensorcnt)
#define HAVE_TDM 1
#else
#define HAVE_TDM 0
#endif
#else
#define HAVE_TDM 0
#endif

__device__ __forceinline__ v8f wmma_bf16(v16bf a, v16bf b, v8f c) {
  // 8 args: (neg_a, A, neg_b, B, c_mod, C, reuse_a, reuse_b)
  return __builtin_amdgcn_wmma_f32_16x16x32_bf16(false, a, false, b, (short)0, c,
                                                 false, false);
}

// Load a 16(row/col) x 32(K) bf16 fragment in the CDNA5 WMMA A/B layout from a
// row-major matrix (leading stride = ld elements). `base` points at (idx0, k0).
// Lane half h holds K in [8h,8h+8) and [16+8h,16+8h+8): two b128 loads.
__device__ __forceinline__ v16bf load_frag16(const __bf16* base, int ld, int lane) {
  const int r = lane & 15;
  const int h = lane >> 4;
  const __bf16* p = base + (size_t)r * ld + (h << 3);
  bf16x8 lo = *(const bf16x8*)(p);
  bf16x8 hi = *(const bf16x8*)(p + 16);
  v16bf f;
#pragma unroll
  for (int i = 0; i < 8; ++i) { f[i] = lo[i]; f[i + 8] = hi[i]; }
  return f;
}

#if HAVE_TDM
// TDM 2D-tile descriptor (CDNA5 ISA ch.8). Group1 (dims/strides/tile shape) is
// built once per kernel and kept loop-invariant in SGPRs; per-issue we patch
// only group0's lds_addr + global_addr dwords.
struct TdmDesc {
  v4u g0;
  v8i g1;
};

__device__ __forceinline__ TdmDesc tdm_make_2d(unsigned tensor_d0, unsigned tensor_d1,
                                               unsigned stride_d0,
                                               unsigned tile_d0, unsigned tile_d1) {
  TdmDesc d;
  d.g0[0] = 1u;          // count=1 (valid), user mode, no gather
  d.g0[1] = 0u;          // lds_addr (patched per issue)
  d.g0[2] = 0u;          // global_addr lo (patched)
  d.g0[3] = 2u << 30;    // type=2 ("image"); global_addr hi patched in
  d.g1[0] = (int)(1u << 16);                          // data_size=1 (2 bytes)
  d.g1[1] = (int)((tensor_d0 & 0xffffu) << 16);       // tensor_dim0[15:0]
  d.g1[2] = (int)(((tensor_d0 >> 16) & 0xffffu) | ((tensor_d1 & 0xffffu) << 16));
  d.g1[3] = (int)(((tensor_d1 >> 16) & 0xffffu) | ((tile_d0 & 0xffffu) << 16));
  d.g1[4] = (int)(tile_d1 & 0xffffu);                 // tile_dim1, tile_dim2=0
  d.g1[5] = (int)stride_d0;                           // tensor_dim0_stride[31:0]
  d.g1[6] = 0;
  d.g1[7] = 0;
  return d;
}

__device__ __forceinline__ void tdm_issue(TdmDesc d, unsigned lds_off,
                                          const void* gptr) {
  const unsigned long long ga = (unsigned long long)(uintptr_t)gptr;
  d.g0[1] = lds_off;
  d.g0[2] = (unsigned)(ga & 0xffffffffu);
  d.g0[3] = (unsigned)((ga >> 32) & 0x1ffffffu) | (2u << 30);
  const v4i z4 = {0, 0, 0, 0};
#if __clang_major__ >= 23
  const v8i z8 = {0, 0, 0, 0, 0, 0, 0, 0};
  __builtin_amdgcn_tensor_load_to_lds(d.g0, d.g1, z4, z4, z8, 0);
#else
  __builtin_amdgcn_tensor_load_to_lds(d.g0, d.g1, z4, z4, 0);
#endif
}
#endif  // HAVE_TDM

// ---------------------------------------------------------------------------
// fp32 -> bf16 conversion (vectorized x4)
// ---------------------------------------------------------------------------
__global__ void cvt_f32_bf16(const float* __restrict__ in, __bf16* __restrict__ out,
                             int n4) {
  int i = blockIdx.x * blockDim.x + threadIdx.x;
  if (i < n4) {
    const float4 v = ((const float4*)in)[i];
    bf16x4 o;
    o[0] = (__bf16)v.x; o[1] = (__bf16)v.y; o[2] = (__bf16)v.z; o[3] = (__bf16)v.w;
    ((bf16x4*)out)[i] = o;
  }
}

// ---------------------------------------------------------------------------
// WMMA GEMM: out = A[M x D] * W^T  (W stored row-major [N x D], i.e. 'od')
// Block tile 128x128, 8 waves; wave (wm=wave&3, wn=wave>>2) owns a 32x64
// quadrant = 2x4 WMMA tiles. K staged through double-buffered LDS via TDM.
// MODE 0: fp32 row-major output (final Wo projection)
// MODE 1: bf16 head-split [b,h,t,d] output (V)
// MODE 2: MODE 1 + fused RoPE (Q, K)
// ---------------------------------------------------------------------------
template <int MODE>
__global__ __launch_bounds__(256)
void proj_gemm(const __bf16* __restrict__ A, const __bf16* __restrict__ W,
               void* __restrict__ out, const int* __restrict__ tokpos) {
  extern __shared__ __bf16 smem[];  // 2 x (A 128x32 | W 128x32) = 32 KB
  const int lane = threadIdx.x & 31;
  const int wave = threadIdx.x >> 5;
  const int wm = wave & 3, wn = wave >> 2;
  const int rowB = blockIdx.y * 128;
  const int colB = blockIdx.x * 128;
  const int row0 = rowB + wm * 32;
  const int col0 = colB + wn * 64;

  const __bf16* Ab = A + (size_t)rowB * D_MODEL;  // tile row band of A
  const __bf16* Wb = W + (size_t)colB * D_MODEL;  // tile row band of W

  v8f acc[2][4] = {};

#if HAVE_TDM
  const bool w0 = (threadIdx.x < 32);
  const TdmDesc dA = tdm_make_2d(D_MODEL, MROWS, D_MODEL, 32u, 128u);
  const TdmDesc dW = tdm_make_2d(D_MODEL, D_MODEL, D_MODEL, 32u, 128u);
  // Pre-issue tile 0 (two TDM ops: A band + W band, each 128 rows x 32 K).
  if (w0) {
    tdm_issue(dA, 0u, Ab);
    tdm_issue(dW, 8192u, Wb);
  }
#pragma unroll 2
  for (int it = 0; it < D_MODEL / 32; ++it) {
    const int cur = it & 1;
    if (w0) {
      if (it + 1 < D_MODEL / 32) {
        const unsigned boff = (unsigned)((it + 1) & 1) * 16384u;
        const int knext = (it + 1) * 32;
        tdm_issue(dA, boff, Ab + knext);
        tdm_issue(dW, boff + 8192u, Wb + knext);
        __builtin_amdgcn_s_wait_tensorcnt(2);  // current pair complete (in-order)
      } else {
        __builtin_amdgcn_s_wait_tensorcnt(0);
      }
    }
    __syncthreads();  // publish buffer `cur` to all waves
    const __bf16* As = smem + cur * 8192;
    const __bf16* Bs = As + 4096;
    v16bf a0 = load_frag16(As + (wm * 32) * 32, 32, lane);
    v16bf a1 = load_frag16(As + (wm * 32 + 16) * 32, 32, lane);
#pragma unroll
    for (int fn = 0; fn < 4; ++fn) {
      v16bf bf = load_frag16(Bs + (wn * 64 + fn * 16) * 32, 32, lane);
      acc[0][fn] = wmma_bf16(a0, bf, acc[0][fn]);
      acc[1][fn] = wmma_bf16(a1, bf, acc[1][fn]);
    }
    __syncthreads();  // all reads of `cur` done before it is re-filled
  }
#else
  // Fallback: direct global fragment loads (pipelined by the scheduler).
  const __bf16* pa0 = A + (size_t)row0 * D_MODEL;
  const __bf16* pa1 = A + (size_t)(row0 + 16) * D_MODEL;
#pragma unroll 2
  for (int k0 = 0; k0 < D_MODEL; k0 += 32) {
    __builtin_prefetch((const void*)(pa0 + k0 + 256), 0, 0);
    v16bf a0 = load_frag16(pa0 + k0, D_MODEL, lane);
    v16bf a1 = load_frag16(pa1 + k0, D_MODEL, lane);
#pragma unroll
    for (int fn = 0; fn < 4; ++fn) {
      v16bf bf = load_frag16(W + (size_t)(col0 + fn * 16) * D_MODEL + k0, D_MODEL, lane);
      acc[0][fn] = wmma_bf16(a0, bf, acc[0][fn]);
      acc[1][fn] = wmma_bf16(a1, bf, acc[1][fn]);
    }
  }
#endif

  // Epilogue. C layout: element (m = r + 8*(lane>>4), n = lane&15) in vgpr r.
  const int r_ = lane & 15, hh = lane >> 4;
#pragma unroll
  for (int fm = 0; fm < 2; ++fm)
#pragma unroll
    for (int fn = 0; fn < 4; ++fn)
#pragma unroll
      for (int r = 0; r < 8; ++r) {
        const int m = row0 + fm * 16 + r + (hh << 3);
        const int n = col0 + fn * 16 + r_;
        float v = acc[fm][fn][r];
        if (MODE == 0) {
          ((float*)out)[(size_t)m * D_MODEL + n] = v;
        } else {
          if (MODE == 2) {
            // RoPE: pair partner is the neighboring column (d^1) -> lane^1.
            const float partner = __shfl_xor(v, 1, 32);
            const int d = n & (DK - 1);
            const int tpos = m & (TT - 1);
            const float pos = (float)tokpos[tpos];
            // inv_freq = theta^(-(d&~1)/DK) = exp(-(d&~1) * ln(1e4)/DK)
            const float freq =
                __expf((float)(d & ~1) * (-9.2103403719761836f / (float)DK));
            float sn, cs;
            __sincosf(pos * freq, &sn, &cs);
            v = (d & 1) ? (v * cs + partner * sn) : (v * cs - partner * sn);
          }
          const int b = m >> 11;           // m / TT
          const int t = m & (TT - 1);
          const int head = n >> 6;         // n / DK
          const int d = n & (DK - 1);
          ((__bf16*)out)[((((size_t)(b * NUM_HEADS + head)) << 11) + t) * DK + d] =
              (__bf16)v;
        }
      }
}

// ---------------------------------------------------------------------------
// Flash-style causal attention. One block per (q-tile of 64 rows, head, batch).
// 4 waves; wave w owns 16 q rows. Streaming softmax over key tiles of 64.
// P is re-laid-out C->A through LDS; V is transposed into LDS for B fragments.
// Output: merged bf16 [B*T, D_MODEL].
// ---------------------------------------------------------------------------
__global__ __launch_bounds__(128)
void attn_fused(const __bf16* __restrict__ Q, const __bf16* __restrict__ K,
                const __bf16* __restrict__ V, __bf16* __restrict__ Om) {
  __shared__ __bf16 Vt[64][72];       // V tile transposed: Vt[d][key], pad 8
  __shared__ __bf16 Ps[4][16][72];    // per-wave P tile (16 q x 64 keys), pad 8

  const int lane = threadIdx.x & 31;
  const int wave = threadIdx.x >> 5;
  const int qt = blockIdx.x;          // 0..31
  const int head = blockIdx.y;        // 0..15
  const int b = blockIdx.z;           // 0..1

  const size_t bh_rows = ((size_t)(b * NUM_HEADS + head)) << 11;  // *2048
  const __bf16* Qp = Q + (bh_rows << 6);
  const __bf16* Kp = K + (bh_rows << 6);
  const __bf16* Vp = V + (bh_rows << 6);

  const int q0 = qt * 64 + wave * 16;   // first q row this wave owns
  const int r_ = lane & 15, hh = lane >> 4;

  // Q fragments (16 rows x 64 d -> two K=32 fragments), loaded once.
  v16bf qf0 = load_frag16(Qp + (size_t)q0 * DK, DK, lane);
  v16bf qf1 = load_frag16(Qp + (size_t)q0 * DK + 32, DK, lane);

  v8f oacc[4] = {};
  float mrow[8], lrow[8];
#pragma unroll
  for (int r = 0; r < 8; ++r) { mrow[r] = -3.0e38f; lrow[r] = 0.0f; }

  for (int kt = 0; kt <= qt; ++kt) {
    const int kbase = kt * 64;

    // Cooperative transposed load of the V tile into LDS.
    {
      const int tid = threadIdx.x;
      const int krow = tid >> 1;            // 0..63
      const int dseg = (tid & 1) * 32;      // 0 or 32
      const __bf16* src = Vp + (size_t)(kbase + krow) * DK + dseg;
#pragma unroll
      for (int j = 0; j < 4; ++j) {
        bf16x8 vv = *(const bf16x8*)(src + j * 8);
#pragma unroll
        for (int e = 0; e < 8; ++e) Vt[dseg + j * 8 + e][krow] = vv[e];
      }
    }
    __syncthreads();

    // S = Q * K^T  (16 q x 64 keys), K fragments contiguous from [t][d] layout.
    v8f sacc[4] = {};
#pragma unroll
    for (int f = 0; f < 4; ++f) {
      v16bf kf0 = load_frag16(Kp + (size_t)(kbase + f * 16) * DK, DK, lane);
      v16bf kf1 = load_frag16(Kp + (size_t)(kbase + f * 16) * DK + 32, DK, lane);
      sacc[f] = wmma_bf16(qf0, kf0, sacc[f]);
      sacc[f] = wmma_bf16(qf1, kf1, sacc[f]);
    }

    // Scale, causal mask, running row max.
    float mnew[8];
#pragma unroll
    for (int r = 0; r < 8; ++r) mnew[r] = mrow[r];
#pragma unroll
    for (int f = 0; f < 4; ++f) {
      const int kcol = kbase + f * 16 + r_;
#pragma unroll
      for (int r = 0; r < 8; ++r) {
        float s = sacc[f][r] * 0.125f;      // 1/sqrt(64)
        const int qrow = q0 + r + (hh << 3);
        if (kcol > qrow) s = -3.0e38f;
        sacc[f][r] = s;
        mnew[r] = fmaxf(mnew[r], s);
      }
    }
#pragma unroll
    for (int mask = 1; mask <= 8; mask <<= 1)
#pragma unroll
      for (int r = 0; r < 8; ++r)
        mnew[r] = fmaxf(mnew[r], __shfl_xor(mnew[r], mask, 32));

    float corr[8], rsum[8];
#pragma unroll
    for (int r = 0; r < 8; ++r) {
      corr[r] = __expf(mrow[r] - mnew[r]);
      mrow[r] = mnew[r];
      lrow[r] *= corr[r];
      rsum[r] = 0.0f;
    }
#pragma unroll
    for (int f = 0; f < 4; ++f)
#pragma unroll
      for (int r = 0; r < 8; ++r) {
        const float p = __expf(sacc[f][r] - mnew[r]);
        sacc[f][r] = p;
        rsum[r] += p;
      }
#pragma unroll
    for (int mask = 1; mask <= 8; mask <<= 1)
#pragma unroll
      for (int r = 0; r < 8; ++r) rsum[r] += __shfl_xor(rsum[r], mask, 32);
#pragma unroll
    for (int r = 0; r < 8; ++r) lrow[r] += rsum[r];

    // Rescale O accumulators by per-row correction.
#pragma unroll
    for (int f = 0; f < 4; ++f)
#pragma unroll
      for (int r = 0; r < 8; ++r) oacc[f][r] *= corr[r];

    // Stage P (C layout) into LDS so it can be reloaded in A-fragment layout.
#pragma unroll
    for (int f = 0; f < 4; ++f)
#pragma unroll
      for (int r = 0; r < 8; ++r)
        Ps[wave][r + (hh << 3)][f * 16 + r_] = (__bf16)sacc[f][r];
    __syncthreads();

    // O += P * V  (K = 64 keys -> two K=32 steps, 4 d-subtiles)
#pragma unroll
    for (int ks = 0; ks < 2; ++ks) {
      v16bf pf = load_frag16(&Ps[wave][0][0] + ks * 32, 72, lane);
#pragma unroll
      for (int f = 0; f < 4; ++f) {
        v16bf vf = load_frag16(&Vt[f * 16][0] + ks * 32, 72, lane);
        oacc[f] = wmma_bf16(pf, vf, oacc[f]);
      }
    }
    __syncthreads();   // protect Vt before next tile overwrites it
  }

  // Normalize and write merged output [B*T, D_MODEL] (heads re-interleaved).
#pragma unroll
  for (int f = 0; f < 4; ++f)
#pragma unroll
    for (int r = 0; r < 8; ++r) {
      const int qrow = q0 + r + (hh << 3);
      const int d = f * 16 + r_;
      const float v = oacc[f][r] / lrow[r];
      Om[(((size_t)(b * TT + qrow)) << 10) + (head << 6) + d] = (__bf16)v;
    }
}

// ---------------------------------------------------------------------------
extern "C" void kernel_launch(void* const* d_in, const int* in_sizes, int n_in,
                              void* d_out, int out_size, void* d_ws, size_t ws_size,
                              hipStream_t stream) {
  (void)in_sizes; (void)n_in; (void)out_size; (void)ws_size;
  const float* x  = (const float*)d_in[0];
  const float* Wq = (const float*)d_in[1];
  const float* Wk = (const float*)d_in[2];
  const float* Wv = (const float*)d_in[3];
  const float* Wo = (const float*)d_in[4];
  const int* tokpos = (const int*)d_in[5];

  const size_t XB = (size_t)MROWS * D_MODEL;     // 4M elements
  const size_t WS = (size_t)D_MODEL * D_MODEL;   // 1M elements
  __bf16* ws  = (__bf16*)d_ws;
  __bf16* xb  = ws;
  __bf16* wqb = xb + XB;
  __bf16* wkb = wqb + WS;
  __bf16* wvb = wkb + WS;
  __bf16* wob = wvb + WS;
  __bf16* Qb  = wob + WS;
  __bf16* Kb  = Qb + XB;
  __bf16* Vb  = Kb + XB;
  __bf16* Mb  = Vb + XB;   // total 48 MB of workspace

  // fp32 -> bf16
  cvt_f32_bf16<<<(int)(XB / 4 / 256), 256, 0, stream>>>(x, xb, (int)(XB / 4));
  cvt_f32_bf16<<<(int)(WS / 4 / 256), 256, 0, stream>>>(Wq, wqb, (int)(WS / 4));
  cvt_f32_bf16<<<(int)(WS / 4 / 256), 256, 0, stream>>>(Wk, wkb, (int)(WS / 4));
  cvt_f32_bf16<<<(int)(WS / 4 / 256), 256, 0, stream>>>(Wv, wvb, (int)(WS / 4));
  cvt_f32_bf16<<<(int)(WS / 4 / 256), 256, 0, stream>>>(Wo, wob, (int)(WS / 4));

  // Projections (+ fused RoPE for Q/K), head-split bf16 outputs.
  dim3 gp(D_MODEL / 128, MROWS / 128), bp(256);
  const size_t lds_bytes = 32768;  // double-buffered 128x32 TDM tiles (A+W)
  proj_gemm<2><<<gp, bp, lds_bytes, stream>>>(xb, wqb, (void*)Qb, tokpos);
  proj_gemm<2><<<gp, bp, lds_bytes, stream>>>(xb, wkb, (void*)Kb, tokpos);
  proj_gemm<1><<<gp, bp, lds_bytes, stream>>>(xb, wvb, (void*)Vb, nullptr);

  // Causal flash attention -> merged bf16.
  attn_fused<<<dim3(TT / 64, NUM_HEADS, BB), 128, 0, stream>>>(Qb, Kb, Vb, Mb);

  // Output projection, fp32 result straight into d_out.
  proj_gemm<0><<<gp, bp, lds_bytes, stream>>>(Mb, wob, d_out, nullptr);
}
